// AttentionBlock_33474975105199
// MI455X (gfx1250) — matrix operand order repaired and checked by reading, not compiled, not run on hardware
//
#include <hip/hip_runtime.h>

#define IN_CH 512
#define HID   64
#define NPIX  4096
#define NB    8

typedef __bf16 bf16_t;
typedef __attribute__((ext_vector_type(16))) __bf16 v16bf;
typedef __attribute__((ext_vector_type(8)))  float  v8f;
typedef __attribute__((ext_vector_type(4)))  unsigned int v4u;
typedef __attribute__((ext_vector_type(8)))  int v8i;
typedef __attribute__((ext_vector_type(4)))  int v4i;

static __device__ __forceinline__ v8f wmma_bf16(v16bf a, v16bf b, v8f c) {
  // D = A(16x32) * B(32x16) + C, f32 accum
  return __builtin_amdgcn_wmma_f32_16x16x32_bf16(false, a, false, b,
                                                 (short)0, c, false, false);
}

// A-fragment K index map (16-bit A 16x32): lanes 0-15 hold K{0..15},
// lanes 16-31 hold K{16..31}; 2 packed values per VGPR.
static __device__ __forceinline__ int a_kmap(int i, int half) {
  return i + half * 16;
}

// ---------------------------------------------------------------------------
// TDM: issue a 2D tensor_load_to_lds (bf16 elements). One call per wave.
// D# group0: count=1, lds_addr, global_addr, type=2. group1: data_size=1(2B),
// tensor_dim0/1, tile_dim0/1, tensor_dim0_stride (ISA 08_async_tensor 8.3-8.4).
// This toolchain exposes the 6-arg builtin (g0, g1, g2, g3, g4, cpol).
// ---------------------------------------------------------------------------
static __device__ __forceinline__ void tdm_load_2d(
    unsigned lds_addr, const void* gptr,
    unsigned tile_d0, unsigned tile_d1,
    unsigned tensor_d0, unsigned tensor_d1, unsigned stride0)
{
  unsigned long long ga = (unsigned long long)(uintptr_t)gptr;
  unsigned ga_lo = (unsigned)__builtin_amdgcn_readfirstlane((int)(ga & 0xffffffffu));
  unsigned ga_hi = (unsigned)__builtin_amdgcn_readfirstlane((int)(ga >> 32));
  unsigned la    = (unsigned)__builtin_amdgcn_readfirstlane((int)lds_addr);

  v4u g0;
  g0[0] = 1u;                                    // count=1, user descriptor
  g0[1] = la;                                    // lds_addr (bytes)
  g0[2] = ga_lo;                                 // global_addr[31:0]
  g0[3] = (ga_hi & 0x01ffffffu) | (2u << 30);    // global_addr[56:32] | type=2

  v8i g1;
  g1[0] = (int)(1u << 16);                       // data_size=1 (2 bytes)
  g1[1] = (int)((tensor_d0 & 0xffffu) << 16);    // tensor_dim0[15:0]
  g1[2] = (int)((tensor_d0 >> 16) & 0xffffu) |   // tensor_dim0[31:16]
          (int)((tensor_d1 & 0xffffu) << 16);    // tensor_dim1[15:0]
  g1[3] = (int)((tensor_d1 >> 16) & 0xffffu) |   // tensor_dim1[31:16]
          (int)(tile_d0 << 16);                  // tile_dim0
  g1[4] = (int)(tile_d1 & 0xffffu);              // tile_dim1 (tile_dim2=0)
  g1[5] = (int)stride0;                          // tensor_dim0_stride[31:0]
  g1[6] = 0;                                     // stride[47:32]=0
  g1[7] = 0;
  v4i z4 = {0, 0, 0, 0};                         // groups 2/3 unused (<=2D)
  v8i z8 = {0, 0, 0, 0, 0, 0, 0, 0};
  __builtin_amdgcn_tensor_load_to_lds(g0, g1, z4, z4, z8, 0);
}

// ---------------------------------------------------------------------------
// Kernel 1: qkv = W_in(192x512) * X(512xN) + b_in
//   -> bf16 Q(scaled) [B][N][64], K [B][N][64], Vt [B][64][N]
// One wave computes a 16(out-ch) x 16(pixel) tile; K=512 via 16 chained WMMAs.
// ---------------------------------------------------------------------------
__global__ __launch_bounds__(256) void qkv_proj_kernel(
    const float* __restrict__ x, const float* __restrict__ w_in,
    const float* __restrict__ b_in,
    bf16_t* __restrict__ Qb, bf16_t* __restrict__ Kb, bf16_t* __restrict__ Vt)
{
  const int lane = threadIdx.x & 31;
  const int wave = threadIdx.x >> 5;
  const int tile = blockIdx.x * 8 + wave;          // 12*256*8 tiles total
  const int b    = tile / (12 * 256);
  const int rem  = tile % (12 * 256);
  const int obase = (rem / 256) * 16;
  const int pbase = (rem % 256) * 16;
  const int half = lane >> 4;
  const int l16  = lane & 15;

  v8f acc = {};
  for (int kb = 0; kb < IN_CH; kb += 32) {
    v16bf a, bm;
#pragma unroll
    for (int i = 0; i < 16; ++i) {
      a[i]  = (bf16_t)w_in[(obase + l16) * IN_CH + kb + a_kmap(i, half)];
      bm[i] = (bf16_t)x[(size_t)(b * IN_CH + kb + half * 16 + i) * NPIX + pbase + l16];
    }
    acc = wmma_bf16(a, bm, acc);
  }

  const float scale = 0.125f;  // 1/sqrt(64), folded into Q
#pragma unroll
  for (int r = 0; r < 8; ++r) {
    const int o = obase + r + half * 8;            // out channel (row)
    const int p = pbase + l16;                     // pixel (col)
    const float v = acc[r] + b_in[o];
    if (o < 64) {
      Qb[((size_t)b * NPIX + p) * HID + o] = (bf16_t)(v * scale);
    } else if (o < 128) {
      Kb[((size_t)b * NPIX + p) * HID + (o - 64)] = (bf16_t)v;
    } else {  // V stored transposed: [B][hid][pixel]
      Vt[((size_t)b * HID + (o - 128)) * NPIX + p] = (bf16_t)v;
    }
  }
}

// ---------------------------------------------------------------------------
// Kernel 2: flash attention. Block = 128 threads (4 waves); each wave owns a
// 16-query tile. Wave 0 drives double-buffered TDM DMA of 32-key K/V tiles
// into LDS; all fragment reads are contiguous 16-element runs (ds_load_b128).
// ---------------------------------------------------------------------------
__global__ __launch_bounds__(128) void attn_kernel(
    const bf16_t* __restrict__ Qb, const bf16_t* __restrict__ Kb,
    const bf16_t* __restrict__ Vt, float* __restrict__ O)
{
  __shared__ bf16_t ldsK[2][32 * 64];   // [key][hid] tiles, 4 KB each
  __shared__ bf16_t ldsVt[2][64 * 32];  // [hid][key] tiles, 4 KB each
  __shared__ bf16_t ldsP[4][16 * 32];   // per-wave P scratch

  const int lane = threadIdx.x & 31;
  const int wave = threadIdx.x >> 5;
  const int half = lane >> 4;
  const int l16  = lane & 15;

  const int qtile = blockIdx.x * 4 + wave;         // 8*256 tiles total
  const int b     = qtile >> 8;
  const int qbase = (qtile & 255) * 16;
  const int bdma  = blockIdx.x >> 6;               // batch for DMA (uniform)

  const bf16_t* Kbase  = Kb + (size_t)bdma * NPIX * HID;
  const bf16_t* Vtbase = Vt + (size_t)bdma * HID * NPIX;
  const unsigned ldsK_off  = (unsigned)(uintptr_t)&ldsK[0][0];
  const unsigned ldsVt_off = (unsigned)(uintptr_t)&ldsVt[0][0];

  // Q fragments (hid chunks 0-31, 32-63); scale already folded in.
  v16bf qf[2];
  {
    const bf16_t* qrow = Qb + ((size_t)b * NPIX + qbase + l16) * HID;
#pragma unroll
    for (int hc = 0; hc < 2; ++hc)
#pragma unroll
      for (int i = 0; i < 16; ++i)
        qf[hc][i] = qrow[hc * 32 + a_kmap(i, half)];
  }

  v8f o0 = {}, o1 = {}, o2 = {}, o3 = {};
  float rowmax[8], rowsum[8];
#pragma unroll
  for (int r = 0; r < 8; ++r) { rowmax[r] = -1e30f; rowsum[r] = 0.f; }

  const int NT = NPIX / 32;  // 128 key tiles

  // Prologue: DMA tile 0 into buffer 0 (wave 0 only; TENSORcnt is per-wave).
  if (wave == 0) {
    // K tile: 32 keys x 64 hid, fully contiguous -> 1 row of 2048 elements
    tdm_load_2d(ldsK_off, Kbase, 2048, 1, 2048, 1, 2048);
    // Vt tile: 64 hid rows x 32 keys, row stride = NPIX elements
    tdm_load_2d(ldsVt_off, Vtbase, 32, 64, NPIX, HID, NPIX);
  }

  for (int kt = 0; kt < NT; ++kt) {
    const int buf = kt & 1;
    if (wave == 0) {
      if (kt + 1 < NT) {  // prefetch next tile into the other buffer
        const int nb = (kt + 1) & 1;
        tdm_load_2d(ldsK_off + (unsigned)nb * 4096,
                    Kbase + (size_t)(kt + 1) * 32 * HID, 2048, 1, 2048, 1, 2048);
        tdm_load_2d(ldsVt_off + (unsigned)nb * 4096,
                    Vtbase + (size_t)(kt + 1) * 32, 32, 64, NPIX, HID, NPIX);
        __builtin_amdgcn_s_wait_tensorcnt(2);  // current pair landed
      } else {
        __builtin_amdgcn_s_wait_tensorcnt(0);
      }
    }
    __syncthreads();  // tile kt visible to all waves

    // S = Q * K^T : two 16-key halves, hid=64 via two chained WMMAs each
    v8f s0 = {}, s1 = {};
#pragma unroll
    for (int hc = 0; hc < 2; ++hc) {
      v16bf bk0, bk1;
#pragma unroll
      for (int i = 0; i < 16; ++i) {
        const int c = hc * 32 + half * 16 + i;     // consecutive in i
        bk0[i] = ldsK[buf][(0 * 16 + l16) * 64 + c];
        bk1[i] = ldsK[buf][(1 * 16 + l16) * 64 + c];
      }
      s0 = wmma_bf16(qf[hc], bk0, s0);
      s1 = wmma_bf16(qf[hc], bk1, s1);
    }

    // Online softmax. C-layout: element r -> row (r + 8*half); col = l16.
    float alpha[8];
#pragma unroll
    for (int r = 0; r < 8; ++r) {
      float m = fmaxf(s0[r], s1[r]);
#pragma unroll
      for (int off = 8; off >= 1; off >>= 1)       // reduce over 16-lane group
        m = fmaxf(m, __shfl_xor(m, off, 32));
      const float newm = fmaxf(rowmax[r], m);
      alpha[r] = __expf(rowmax[r] - newm);
      rowmax[r] = newm;
      const float p0 = __expf(s0[r] - newm);
      const float p1 = __expf(s1[r] - newm);
      s0[r] = p0; s1[r] = p1;
      float ls = p0 + p1;
#pragma unroll
      for (int off = 8; off >= 1; off >>= 1)
        ls += __shfl_xor(ls, off, 32);
      rowsum[r] = rowsum[r] * alpha[r] + ls;
      o0[r] *= alpha[r]; o1[r] *= alpha[r];
      o2[r] *= alpha[r]; o3[r] *= alpha[r];
    }

    // P (C-layout) -> LDS -> A-layout fragment
    bf16_t* P = ldsP[wave];
#pragma unroll
    for (int r = 0; r < 8; ++r) {
      const int row = r + half * 8;
      P[row * 32 + l16]      = (bf16_t)s0[r];
      P[row * 32 + 16 + l16] = (bf16_t)s1[r];
    }
    __syncthreads();

    v16bf pf;
#pragma unroll
    for (int i = 0; i < 16; ++i)
      pf[i] = P[l16 * 32 + a_kmap(i, half)];       // consecutive in i

    // O(16x64) += P(16x32) * V(32x64): four N-chunks of 16.
    // Vt tile is [hid][key] so each fragment is a contiguous 16-element run.
#pragma unroll
    for (int nc = 0; nc < 4; ++nc) {
      v16bf vf;
#pragma unroll
      for (int i = 0; i < 16; ++i)
        vf[i] = ldsVt[buf][(nc * 16 + l16) * 32 + half * 16 + i];
      switch (nc) {
        case 0: o0 = wmma_bf16(pf, vf, o0); break;
        case 1: o1 = wmma_bf16(pf, vf, o1); break;
        case 2: o2 = wmma_bf16(pf, vf, o2); break;
        default: o3 = wmma_bf16(pf, vf, o3); break;
      }
    }
    __syncthreads();  // all LDS reads of tile kt done before buffer reuse
  }

  // Normalize and write O [B][N][64] fp32
#pragma unroll
  for (int r = 0; r < 8; ++r) {
    const int row = qbase + r + half * 8;
    const float inv = 1.0f / rowsum[r];
    float* orow = O + ((size_t)b * NPIX + row) * HID;
    orow[0 * 16 + l16] = o0[r] * inv;
    orow[1 * 16 + l16] = o1[r] * inv;
    orow[2 * 16 + l16] = o2[r] * inv;
    orow[3 * 16 + l16] = o3[r] * inv;
  }
}

// ---------------------------------------------------------------------------
// Kernel 3: y = W_out(512x64) * O^T + b_out + x  (identity skip), fp32 out
// ---------------------------------------------------------------------------
__global__ __launch_bounds__(256) void out_proj_kernel(
    const float* __restrict__ Obuf, const float* __restrict__ w_out,
    const float* __restrict__ b_out, const float* __restrict__ x,
    float* __restrict__ y)
{
  const int lane = threadIdx.x & 31;
  const int wave = threadIdx.x >> 5;
  const int tile = blockIdx.x * 8 + wave;          // 32*256*8 tiles
  const int b    = tile / (32 * 256);
  const int rem  = tile % (32 * 256);
  const int obase = (rem / 256) * 16;
  const int pbase = (rem % 256) * 16;
  const int half = lane >> 4;
  const int l16  = lane & 15;

  v8f acc = {};
#pragma unroll
  for (int hc = 0; hc < 2; ++hc) {
    v16bf a, bm;
#pragma unroll
    for (int i = 0; i < 16; ++i) {
      a[i]  = (bf16_t)w_out[(obase + l16) * HID + hc * 32 + a_kmap(i, half)];
      bm[i] = (bf16_t)Obuf[((size_t)b * NPIX + pbase + l16) * HID
                           + hc * 32 + half * 16 + i];
    }
    acc = wmma_bf16(a, bm, acc);
  }

#pragma unroll
  for (int r = 0; r < 8; ++r) {
    const int o = obase + r + half * 8;
    const int p = pbase + l16;
    const size_t xi = ((size_t)b * IN_CH + o) * NPIX + p;
    y[xi] = acc[r] + b_out[o] + x[xi];
  }
}

// ---------------------------------------------------------------------------
extern "C" void kernel_launch(void* const* d_in, const int* in_sizes, int n_in,
                              void* d_out, int out_size, void* d_ws, size_t ws_size,
                              hipStream_t stream) {
  const float* x     = (const float*)d_in[0];
  const float* w_in  = (const float*)d_in[1];
  const float* b_in  = (const float*)d_in[2];
  const float* w_out = (const float*)d_in[3];
  const float* b_out = (const float*)d_in[4];
  float* y = (float*)d_out;

  const size_t qkv_elems = (size_t)NB * NPIX * HID;   // 2M elems
  bf16_t* Qb = (bf16_t*)d_ws;
  bf16_t* Kb = Qb + qkv_elems;
  bf16_t* Vt = Kb + qkv_elems;
  float*  Ob = (float*)(Vt + qkv_elems);              // 8 MiB; total ws 20 MiB

  // 1) QKV projection: 12*256*8 = 24576 wave-tiles, 8 waves/block
  qkv_proj_kernel<<<3072, 256, 0, stream>>>(x, w_in, b_in, Qb, Kb, Vt);
  // 2) Flash attention: 8*256 = 2048 query tiles, 4 waves/block
  attn_kernel<<<512, 128, 0, stream>>>(Qb, Kb, Vt, Ob);
  // 3) Output projection + skip: 32*256*8 = 65536 tiles, 8 waves/block
  out_proj_kernel<<<8192, 256, 0, stream>>>(Ob, w_out, b_out, x, y);
}
